// GroupedQueryAttention_11879879544216
// MI455X (gfx1250) — matrix operand order repaired and checked by reading, not compiled, not run on hardware
//
#include <hip/hip_runtime.h>
#include <hip/hip_bf16.h>

// GQA forward on MI455X (gfx1250), bf16 WMMA pipeline + async global->LDS.
// d_ws layout (20 MB needed):
//   [0,8MB)   Q  bf16 (2048 x 2048) row-major, RoPE applied
//   [8,10MB)  K  bf16 (2048 x 512)  row-major, RoPE applied
//   [10,12MB) V  bf16 (2048 x 512)  row-major
//   [12,20MB) attn-out bf16 (2048 x 2048) row-major

typedef __attribute__((ext_vector_type(16))) __bf16 v16bf;
typedef __attribute__((ext_vector_type(8)))  __bf16 v8bf;
typedef __attribute__((ext_vector_type(8)))  float  v8f;
typedef __attribute__((ext_vector_type(4)))  int    v4i;

#define DEV __device__ __forceinline__

// CDNA5 async global->LDS path (ASYNCcnt-tracked), guarded so the file still
// compiles (with the VGPR staging fallback) on toolchains without the builtins.
#if defined(__gfx1250__) &&                                              \
    __has_builtin(__builtin_amdgcn_global_load_async_to_lds_b128) &&     \
    __has_builtin(__builtin_amdgcn_s_wait_asynccnt)
#define USE_ASYNC_LDS 1
#else
#define USE_ASYNC_LDS 0
#endif

#if USE_ASYNC_LDS
// Builtin prototype (from the round-2 diagnostic): src is int4* in AS(1),
// dst is int4* in AS(3), then imm offset, imm cpol.
typedef __attribute__((address_space(1))) v4i* glb_b128p;
typedef __attribute__((address_space(3))) v4i* lds_b128p;
// Generic->AS casts via integers: low 32 bits of a generic LDS pointer are the
// wave-relative LDS byte offset (aperture rule), which is what VDST expects.
DEV void async_cp_b128(const void* g, void* l) {
  __builtin_amdgcn_global_load_async_to_lds_b128(
      (glb_b128p)(unsigned long long)g,
      (lds_b128p)(unsigned int)(unsigned long long)l, 0, 0);
}
DEV void async_wait0() { __builtin_amdgcn_s_wait_asynccnt(0); }
#endif

DEV __bf16 f2bf(float f) {
  unsigned u = __builtin_bit_cast(unsigned, f);
  u += 0x7FFFu + ((u >> 16) & 1u);                 // round-to-nearest-even
  return __builtin_bit_cast(__bf16, (unsigned short)(u >> 16));
}

// Two 16B LDS loads -> one 16-element bf16 fragment register group.
DEV v16bf ld2x8(const __bf16* p, int step) {
  v8bf lo = *(const v8bf*)p;
  v8bf hi = *(const v8bf*)(p + step);
  return __builtin_shufflevector(lo, hi, 0,1,2,3,4,5,6,7,8,9,10,11,12,13,14,15);
}

DEV v8f wmma_bf16(v16bf a, v16bf b, v8f c) {
  // v_wmma_f32_16x16x32_bf16 (neg_a, A, neg_b, B, c_mod, C, reuse_a, reuse_b)
  return __builtin_amdgcn_wmma_f32_16x16x32_bf16(false, a, false, b, (short)0, c,
                                                 false, false);
}

// ---------------------------------------------------------------------------
// GEMM: Out(MxN) = A(MxK, row-major, fp32 or bf16) * Bw(KxN, fp32)
// Block = 256 thr (8 waves), tile 128x64, BK=32. Wave grid 4x2, 32x32 C/wave.
// Optional interleaved RoPE on output columns (per 64-wide head).
// bf16-A tiles are staged with GLOBAL_LOAD_ASYNC_TO_LDS when available.
// ---------------------------------------------------------------------------
template<bool A_BF16, bool ROPE, bool OUT_BF16>
__global__ __launch_bounds__(256)
void gemm_wmma(const void* __restrict__ Av, const float* __restrict__ Bw,
               void* __restrict__ Out, int M, int N, int K) {
  constexpr int BM = 128, BN = 64, BK = 32;
  constexpr int AST = 56;  // 112B pitch: 16B aligned, conflict-free mod 64 banks
  constexpr int BST = 56;
  __shared__ __align__(16) __bf16 lA[BM * AST];     // [row][k]
  __shared__ __align__(16) __bf16 lB[BN * BST];     // transposed: [n][k]

  const int tid  = threadIdx.x;
  const int lane = tid & 31;
  const int wave = tid >> 5;
  const int wm   = wave & 3;          // 0..3 -> 32-row strip
  const int wn   = wave >> 2;         // 0..1 -> 32-col strip
  const int bm   = blockIdx.y * BM;
  const int bn   = blockIdx.x * BN;
  const int lr   = lane & 15;
  const int hi8  = (lane >> 4) * 8;

  v8f acc[2][2];
#pragma unroll
  for (int a = 0; a < 2; ++a)
#pragma unroll
    for (int b = 0; b < 2; ++b)
#pragma unroll
      for (int i = 0; i < 8; ++i) acc[a][b][i] = 0.0f;

  for (int k0 = 0; k0 < K; k0 += BK) {
    __syncthreads();
    // ---- stage A tile (128x32) into LDS as bf16 -------------------------
    if constexpr (A_BF16) {
#if USE_ASYNC_LDS
      // 512 chunks of 8 bf16 (16B) -> one async b128 per chunk, no VGPR hop.
#pragma unroll
      for (int it = 0; it < 2; ++it) {
        int idx = tid + it * 256;          // 0..511
        int r   = idx >> 2;
        int c8  = (idx & 3) * 8;
        async_cp_b128((const __bf16*)Av + (size_t)(bm + r) * K + k0 + c8,
                      &lA[r * AST + c8]);
      }
#else
#pragma unroll
      for (int it = 0; it < 4; ++it) {
        int idx = tid + it * 256;          // 0..1023 -> 4-element chunks
        int r   = idx >> 3;
        int c4  = (idx & 7) * 4;
        const __bf16* src = (const __bf16*)Av + (size_t)(bm + r) * K + k0 + c4;
        if (k0 + BK < K) __builtin_prefetch(src + BK, 0, 0);
        *(uint2*)&lA[r * AST + c4] = *(const uint2*)src;
      }
#endif
    } else {
#pragma unroll
      for (int it = 0; it < 4; ++it) {
        int idx = tid + it * 256;
        int r   = idx >> 3;
        int c4  = (idx & 7) * 4;
        const float* src = (const float*)Av + (size_t)(bm + r) * K + k0 + c4;
        if (k0 + BK < K) __builtin_prefetch(src + BK, 0, 0);  // global_prefetch_b8
        float4 v = *(const float4*)src;
        __bf16* d = &lA[r * AST + c4];
        d[0] = f2bf(v.x); d[1] = f2bf(v.y); d[2] = f2bf(v.z); d[3] = f2bf(v.w);
      }
    }
    // ---- stage B tile (32x64) transposed into LDS (fp32 -> bf16) --------
#pragma unroll
    for (int it = 0; it < 2; ++it) {
      int idx = tid + it * 256;            // 0..511
      int k   = idx >> 4;
      int n4  = (idx & 15) * 4;
      const float* src = Bw + (size_t)(k0 + k) * N + bn + n4;
      if (k0 + BK < K) __builtin_prefetch(src + (size_t)BK * N, 0, 0);
      float4 v = *(const float4*)src;
      lB[(n4 + 0) * BST + k] = f2bf(v.x);
      lB[(n4 + 1) * BST + k] = f2bf(v.y);
      lB[(n4 + 2) * BST + k] = f2bf(v.z);
      lB[(n4 + 3) * BST + k] = f2bf(v.w);
    }
#if USE_ASYNC_LDS
    if constexpr (A_BF16) async_wait0();   // s_wait_asynccnt 0
#endif
    __syncthreads();
    // ---- WMMA compute ---------------------------------------------------
    const int ah = (lane >> 4) * 8;        // A frag: K offsets {0|8}, +16
    const int bh = (lane >> 4) * 8;        // B frag: K offsets {0|8}, +16
    v16bf a0 = ld2x8(&lA[(wm * 32 +      lr) * AST + ah], 16);
    v16bf a1 = ld2x8(&lA[(wm * 32 + 16 + lr) * AST + ah], 16);
    v16bf b0 = ld2x8(&lB[(wn * 32 +      lr) * BST + bh], 16);
    v16bf b1 = ld2x8(&lB[(wn * 32 + 16 + lr) * BST + bh], 16);
    acc[0][0] = wmma_bf16(a0, b0, acc[0][0]);
    acc[0][1] = wmma_bf16(a0, b1, acc[0][1]);
    acc[1][0] = wmma_bf16(a1, b0, acc[1][0]);
    acc[1][1] = wmma_bf16(a1, b1, acc[1][1]);
  }

  // ---- epilogue: optional RoPE, store -----------------------------------
#pragma unroll
  for (int ti = 0; ti < 2; ++ti)
#pragma unroll
    for (int tj = 0; tj < 2; ++tj) {
      v8f c = acc[ti][tj];
      const int col = bn + wn * 32 + tj * 16 + lr;
      float invf = 0.0f;
      if constexpr (ROPE) {
        int ip = (col & 63) >> 1;                       // pair index in head
        invf = __powf(500000.0f, -(float)(2 * ip) * (1.0f / 64.0f));
      }
#pragma unroll
      for (int i = 0; i < 8; ++i) {
        int row = bm + wm * 32 + ti * 16 + i + hi8;     // token index
        float x = c[i];
        if constexpr (ROPE) {
          float xp = __shfl_xor(x, 1, 32);              // even<->odd partner
          float sn, cs;
          __sincosf((float)row * invf, &sn, &cs);
          x = ((col & 1) == 0) ? (x * cs - xp * sn) : (xp * sn + x * cs);
        }
        if constexpr (OUT_BF16)
          ((__bf16*)Out)[(size_t)row * N + col] = f2bf(x);
        else
          ((float*)Out)[(size_t)row * N + col] = x;
      }
    }
}

// ---------------------------------------------------------------------------
// Flash attention, causal, GQA (4 Q-heads per KV head).
// Grid: (qblocks=32, heads=32). Block: 128 thr = 4 waves; wave w owns
// query rows [qb*64 + w*16, +16). Online softmax in WMMA C-layout; row
// reductions via shfl_xor over the 16-lane half-groups. Q/K tiles staged via
// async global->LDS when available; V is transposed through VGPRs.
// ---------------------------------------------------------------------------
__global__ __launch_bounds__(128)
void gqa_flash(const void* __restrict__ Qv, const void* __restrict__ Kv,
               const void* __restrict__ Vv, void* __restrict__ Ov) {
  constexpr int ST = 72;                  // 144B pitch, conflict-free
  __shared__ __align__(16) __bf16 lQ [64 * ST];   // [q][d]
  __shared__ __align__(16) __bf16 lK [64 * ST];   // [kv][d]
  __shared__ __align__(16) __bf16 lVt[64 * ST];   // [d][kv] (transposed)
  __shared__ __align__(16) __bf16 lP [64 * ST];   // [q][kv]

  const __bf16* Q = (const __bf16*)Qv;
  const __bf16* K = (const __bf16*)Kv;
  const __bf16* V = (const __bf16*)Vv;
  __bf16*       O = (__bf16*)Ov;

  const int tid  = threadIdx.x;
  const int lane = tid & 31;
  const int wave = tid >> 5;
  const int qb   = blockIdx.x;
  const int h    = blockIdx.y;
  const int hkv  = h >> 2;                // 32 Q heads / 8 KV heads
  const int lr   = lane & 15;
  const int hi8  = (lane >> 4) * 8;
  const int ah   = (lane >> 4) * 8;       // A-frag K offset
  const int bh   = (lane >> 4) * 8;       // B-frag K offset

  // stage Q tile (64 x 64) once
#if USE_ASYNC_LDS
#pragma unroll
  for (int it = 0; it < 4; ++it) {
    int idx = tid + it * 128;             // 0..511 8-elt (16B) chunks
    int r = idx >> 3, c8 = (idx & 7) * 8;
    async_cp_b128(&Q[(size_t)(qb * 64 + r) * 2048 + h * 64 + c8],
                  &lQ[r * ST + c8]);
  }
#else
#pragma unroll
  for (int it = 0; it < 8; ++it) {
    int idx = tid + it * 128;             // 0..1023 4-elt chunks
    int r = idx >> 4, c4 = (idx & 15) * 4;
    *(uint2*)&lQ[r * ST + c4] =
        *(const uint2*)&Q[(size_t)(qb * 64 + r) * 2048 + h * 64 + c4];
  }
#endif

  float m[8], l[8];
  v8f o[4];
#pragma unroll
  for (int i = 0; i < 8; ++i) { m[i] = -3.0e38f; l[i] = 0.0f; }
#pragma unroll
  for (int j = 0; j < 4; ++j)
#pragma unroll
    for (int i = 0; i < 8; ++i) o[j][i] = 0.0f;

  for (int kb = 0; kb <= qb; ++kb) {      // causal: only lower KV blocks
    __syncthreads();
    // stage K (row-major, async when possible) and V (transposed, via VGPRs)
#if USE_ASYNC_LDS
#pragma unroll
    for (int it = 0; it < 4; ++it) {
      int idx = tid + it * 128;
      int r = idx >> 3, c8 = (idx & 7) * 8;
      async_cp_b128(&K[(size_t)(kb * 64 + r) * 512 + hkv * 64 + c8],
                    &lK[r * ST + c8]);
    }
#pragma unroll
    for (int it = 0; it < 8; ++it) {
      int idx = tid + it * 128;
      int r = idx >> 4, c4 = (idx & 15) * 4;
      uint2 vv = *(const uint2*)&V[(size_t)(kb * 64 + r) * 512 + hkv * 64 + c4];
      const __bf16* pv = (const __bf16*)&vv;
      lVt[(c4 + 0) * ST + r] = pv[0];
      lVt[(c4 + 1) * ST + r] = pv[1];
      lVt[(c4 + 2) * ST + r] = pv[2];
      lVt[(c4 + 3) * ST + r] = pv[3];
    }
    async_wait0();                         // s_wait_asynccnt 0 (covers Q too)
#else
#pragma unroll
    for (int it = 0; it < 8; ++it) {
      int idx = tid + it * 128;
      int r = idx >> 4, c4 = (idx & 15) * 4;
      size_t g = (size_t)(kb * 64 + r) * 512 + hkv * 64 + c4;
      *(uint2*)&lK[r * ST + c4] = *(const uint2*)&K[g];
      uint2 vv = *(const uint2*)&V[g];
      const __bf16* pv = (const __bf16*)&vv;
      lVt[(c4 + 0) * ST + r] = pv[0];
      lVt[(c4 + 1) * ST + r] = pv[1];
      lVt[(c4 + 2) * ST + r] = pv[2];
      lVt[(c4 + 3) * ST + r] = pv[3];
    }
#endif
    __syncthreads();

    // S = (Q K^T) * rsqrt(64): 4 col-tiles x 2 d-steps of WMMA
    v16bf qa0 = ld2x8(&lQ[(wave * 16 + lr) * ST + ah], 16);
    v16bf qa1 = ld2x8(&lQ[(wave * 16 + lr) * ST + 32 + ah], 16);
    v8f s[4];
#pragma unroll
    for (int j = 0; j < 4; ++j) {
      v16bf k0 = ld2x8(&lK[(j * 16 + lr) * ST + bh], 16);
      v16bf k1 = ld2x8(&lK[(j * 16 + lr) * ST + 32 + bh], 16);
      v8f z;
#pragma unroll
      for (int i = 0; i < 8; ++i) z[i] = 0.0f;
      z = wmma_bf16(qa0, k0, z);
      z = wmma_bf16(qa1, k1, z);
      s[j] = z * 0.125f;
    }
    if (kb == qb) {                        // diagonal block: causal mask
#pragma unroll
      for (int j = 0; j < 4; ++j)
#pragma unroll
        for (int i = 0; i < 8; ++i) {
          int col = j * 16 + lr, row = wave * 16 + i + hi8;
          s[j][i] = (col > row) ? -3.0e38f : s[j][i];
        }
    }
    // online softmax: row max/sum across the 16-lane half-groups
    float mc[8];
#pragma unroll
    for (int i = 0; i < 8; ++i)
      mc[i] = fmaxf(fmaxf(s[0][i], s[1][i]), fmaxf(s[2][i], s[3][i]));
#pragma unroll
    for (int off = 1; off < 16; off <<= 1)
#pragma unroll
      for (int i = 0; i < 8; ++i) mc[i] = fmaxf(mc[i], __shfl_xor(mc[i], off, 32));
    float sc[8], rs[8];
#pragma unroll
    for (int i = 0; i < 8; ++i) {
      float mn = fmaxf(m[i], mc[i]);
      sc[i] = __expf(m[i] - mn);
      m[i]  = mn;
      rs[i] = 0.0f;
    }
#pragma unroll
    for (int j = 0; j < 4; ++j)
#pragma unroll
      for (int i = 0; i < 8; ++i) {
        float p = __expf(s[j][i] - m[i]);
        rs[i] += p;
        lP[(wave * 16 + i + hi8) * ST + j * 16 + lr] = f2bf(p);   // C->A via LDS
      }
#pragma unroll
    for (int off = 1; off < 16; off <<= 1)
#pragma unroll
      for (int i = 0; i < 8; ++i) rs[i] += __shfl_xor(rs[i], off, 32);
#pragma unroll
    for (int i = 0; i < 8; ++i) l[i] = l[i] * sc[i] + rs[i];
#pragma unroll
    for (int j = 0; j < 4; ++j)
#pragma unroll
      for (int i = 0; i < 8; ++i) o[j][i] *= sc[i];
    __syncthreads();

    // O += P * V : 2 kv-steps x 4 d-tiles of WMMA
#pragma unroll
    for (int kk = 0; kk < 2; ++kk) {
      v16bf pa = ld2x8(&lP[(wave * 16 + lr) * ST + kk * 32 + ah], 16);
#pragma unroll
      for (int j = 0; j < 4; ++j) {
        v16bf vb = ld2x8(&lVt[(j * 16 + lr) * ST + kk * 32 + bh], 16);
        o[j] = wmma_bf16(pa, vb, o[j]);
      }
    }
  }

  // normalize and store attention output (row-major, (T, 2048))
#pragma unroll
  for (int j = 0; j < 4; ++j)
#pragma unroll
    for (int i = 0; i < 8; ++i) {
      int row = qb * 64 + wave * 16 + i + hi8;
      int col = h * 64 + j * 16 + lr;
      O[(size_t)row * 2048 + col] = f2bf(o[j][i] / l[i]);
    }
}

// ---------------------------------------------------------------------------
extern "C" void kernel_launch(void* const* d_in, const int* in_sizes, int n_in,
                              void* d_out, int out_size, void* d_ws, size_t ws_size,
                              hipStream_t stream) {
  const float* q_embs = (const float*)d_in[0];
  const float* k_embs = (const float*)d_in[1];
  const float* v_embs = (const float*)d_in[2];
  const float* w_q    = (const float*)d_in[3];
  const float* w_k    = (const float*)d_in[4];
  const float* w_v    = (const float*)d_in[5];
  const float* w_o    = (const float*)d_in[6];

  char* ws = (char*)d_ws;
  void* Qws = ws;                                   //  8 MB
  void* Kws = ws + (size_t)8  * 1024 * 1024;        //  2 MB
  void* Vws = ws + (size_t)10 * 1024 * 1024;        //  2 MB
  void* Aws = ws + (size_t)12 * 1024 * 1024;        //  8 MB

  // Projections (fp32 in -> bf16 out), RoPE fused for Q and K.
  gemm_wmma<false, true,  true><<<dim3(32, 16), 256, 0, stream>>>(q_embs, w_q, Qws, 2048, 2048, 2048);
  gemm_wmma<false, true,  true><<<dim3(8,  16), 256, 0, stream>>>(k_embs, w_k, Kws, 2048, 512,  2048);
  gemm_wmma<false, false, true><<<dim3(8,  16), 256, 0, stream>>>(v_embs, w_v, Vws, 2048, 512,  2048);

  // Causal GQA flash attention.
  gqa_flash<<<dim3(32, 32), 128, 0, stream>>>(Qws, Kws, Vws, Aws);

  // Output projection (bf16 x fp32-weights -> fp32 out).
  gemm_wmma<true, false, false><<<dim3(32, 16), 256, 0, stream>>>(Aws, w_o, d_out, 2048, 2048, 2048);
}
